// ResidualVQ_8976481648792
// MI455X (gfx1250) — matrix operand order repaired and checked by reading, not compiled, not run on hardware
//
#include <hip/hip_runtime.h>

// ---------------------------------------------------------------------------
// Residual VQ forward for MI455X (gfx1250).
// B=65536, D=256, R=8, K=1024.
// GEMM path: V_WMMA_F32_16X16X32_BF16 (fp32 accumulate), residual math fp32.
// Argmin: per-lane running min across code tiles (branchless), single
// ds_bpermute butterfly per row at the end of each stage.
// ---------------------------------------------------------------------------

typedef __attribute__((ext_vector_type(16))) __bf16 v16bf;
typedef __attribute__((ext_vector_type(8)))  __bf16 v8bf;
typedef __attribute__((ext_vector_type(8)))  float  v8f;
typedef __attribute__((ext_vector_type(4)))  float  v4f;

#define B_N     65536
#define D_N     256
#define R_N     8
#define K_N     1024
#define TILE_B  64
#define THREADS 256

// d_out flat layout (reference return order): codes[B,R], z_q[B,D],
// residual[B,D], recon_loss, usage[R,K]
#define OFF_CODES 0ull
#define OFF_ZQ    524288ull            // 65536*8
#define OFF_RES   17301504ull          // +65536*256
#define OFF_LOSS  34078720ull          // +65536*256
#define OFF_USAGE 34078721ull

__device__ __forceinline__ float shfl_xor_f(float v, int mask, int lane) {
    int r = __builtin_amdgcn_ds_bpermute(((lane ^ mask) << 2),
                                         __builtin_bit_cast(int, v));
    return __builtin_bit_cast(float, r);
}
__device__ __forceinline__ int shfl_xor_i(int v, int mask, int lane) {
    return __builtin_amdgcn_ds_bpermute(((lane ^ mask) << 2), v);
}

// ---------------------------------------------------------------------------
__global__ void rvq_zero(float* __restrict__ counts, float* __restrict__ lossp) {
    int i = blockIdx.x * blockDim.x + threadIdx.x;
    if (i < R_N * K_N) counts[i] = 0.0f;
    if (i == 0) *lossp = 0.0f;
}

__global__ void rvq_prep_bf16(const float* __restrict__ embed,
                              __bf16* __restrict__ cb16) {
    int total = R_N * K_N * D_N;
    for (int i = blockIdx.x * blockDim.x + threadIdx.x; i < total;
         i += gridDim.x * blockDim.x)
        cb16[i] = (__bf16)embed[i];
}

__global__ void rvq_prep_e2(const float* __restrict__ embed,
                            float* __restrict__ e2) {
    int row = blockIdx.x * blockDim.x + threadIdx.x;
    if (row >= R_N * K_N) return;
    const float* p = embed + (size_t)row * D_N;
    float s = 0.0f;
    #pragma unroll 8
    for (int c = 0; c < D_N / 4; ++c) {
        v4f v = *(const v4f*)(p + c * 4);
        s += v[0] * v[0] + v[1] * v[1] + v[2] * v[2] + v[3] * v[3];
    }
    e2[row] = s;
}

// ---------------------------------------------------------------------------
__global__ __launch_bounds__(THREADS) void rvq_main(
    const float* __restrict__ z, const float* __restrict__ embed,
    const __bf16* __restrict__ cb16, const float* __restrict__ e2,
    float* __restrict__ counts, float* __restrict__ lossp,
    float* __restrict__ out)
{
    // LDS: bf16 residual tile (row stride 264 halves = 528B -> row-to-row
    // bank shift of 4, conflict-free 16-row reads), plus argmin scratch.
    __shared__ __bf16 abuf[TILE_B][264];
    __shared__ float  bd[TILE_B][2];
    __shared__ int    bi[TILE_B][2];
    __shared__ int    codes_sh[TILE_B];

    const int t    = threadIdx.x;
    const int lane = t & 31;
    const int wave = t >> 5;
    const int rowT = t >> 2;              // residual-ownership row (0..63)
    const int q    = t & 3;               // d-quarter [64q, 64q+64)
    const int rowBase = (wave >> 1) * 16; // GEMM row-tile base
    const int colHalf = wave & 1;         // code range half (512 codes)
    const int m  = lane & 15;
    const int hi = lane >> 4;
    const int blockRow0 = blockIdx.x * TILE_B;

    // fp32 residual carried in VGPRs for all 8 stages
    float r_reg[64];
    {
        const float* zp = z + (size_t)(blockRow0 + rowT) * D_N + q * 64;
        #pragma unroll
        for (int c = 0; c < 16; ++c) {
            v4f v = *(const v4f*)(zp + c * 4);
            r_reg[c * 4 + 0] = v[0]; r_reg[c * 4 + 1] = v[1];
            r_reg[c * 4 + 2] = v[2]; r_reg[c * 4 + 3] = v[3];
        }
    }

    for (int s = 0; s < R_N; ++s) {
        // ---- 1. residual (fp32 regs) -> bf16 LDS, row-major -------------
        {
            __bf16* dst = &abuf[rowT][q * 64];
            #pragma unroll
            for (int c = 0; c < 8; ++c) {
                v8bf vv;
                #pragma unroll
                for (int j = 0; j < 8; ++j) vv[j] = (__bf16)r_reg[c * 8 + j];
                *(v8bf*)(dst + c * 8) = vv;
            }
        }
        __syncthreads();

        // ---- 2. WMMA GEMM + per-lane running argmin ---------------------
        // Lane slot m sees candidate indices n0+m, strictly increasing in
        // nt, so strict '<' keeps the earliest (smallest) index on ties.
        float best_d[8]; int best_i[8];
        #pragma unroll
        for (int r = 0; r < 8; ++r) { best_d[r] = 3.4e38f; best_i[r] = 0; }

        const __bf16* cbS = cb16 + (size_t)s * K_N * D_N;
        const float*  e2S = e2 + s * K_N;
        const __bf16* arow = &abuf[rowBase + m][0];

        for (int nt = 0; nt < 32; ++nt) {
            const int n0 = colHalf * 512 + nt * 16;
            // pull the slice of the code tile two ahead (n0+32) toward the WGP caches
            {
                const int pn = (nt >= 30) ? n0 : (n0 + 32);
                __builtin_prefetch(cbS + (size_t)(pn + m) * D_N + hi * 16, 0, 3);
            }
            const float e2v = e2S[n0 + m];
            v8f acc = {0.f, 0.f, 0.f, 0.f, 0.f, 0.f, 0.f, 0.f};
            #pragma unroll
            for (int dk = 0; dk < 8; ++dk) {
                const int d0 = dk * 32;
                // A operand (contiguous k order, not AMD's table order): lane m = row, hi selects
                // K 16..31: element s = K[hi*16 + s] (matches B)
                v8bf alo = *(const v8bf*)(arow + d0 + hi * 16);
                v8bf ahi = *(const v8bf*)(arow + d0 + hi * 16 + 8);
                v16bf a;
                #pragma unroll
                for (int j = 0; j < 8; ++j) { a[j] = alo[j]; a[8 + j] = ahi[j]; }
                // B operand (32x16): lane m = column (code), hi selects
                // K 16..31 -> 16 contiguous bf16 of the codebook row.
                const __bf16* bp = cbS + (size_t)(n0 + m) * D_N + d0 + hi * 16;
                v16bf b = *(const v16bf*)bp;
                acc = __builtin_amdgcn_wmma_f32_16x16x32_bf16(
                        false, a, false, b, (short)0, acc, false, false);
            }
            // dist = e2[k] - 2*dot; branchless running min (v_cndmask)
            const int iv = n0 + m;
            #pragma unroll
            for (int r = 0; r < 8; ++r) {
                float dv = __builtin_fmaf(-2.0f, acc[r], e2v);
                bool better = dv < best_d[r];
                best_d[r] = better ? dv : best_d[r];
                best_i[r] = better ? iv : best_i[r];
            }
        }

        // one butterfly per row across the 16 column slots (branchless)
        #pragma unroll
        for (int r = 0; r < 8; ++r) {
            float dv = best_d[r]; int iv = best_i[r];
            #pragma unroll
            for (int off = 1; off < 16; off <<= 1) {
                float od = shfl_xor_f(dv, off, lane);
                int   oi = shfl_xor_i(iv, off, lane);
                bool take = (od < dv) | ((od == dv) & (oi < iv));
                dv = take ? od : dv;
                iv = take ? oi : iv;
            }
            best_d[r] = dv; best_i[r] = iv;
        }
        if (m == 0) {
            #pragma unroll
            for (int r = 0; r < 8; ++r) {
                int row = rowBase + hi * 8 + r;
                bd[row][colHalf] = best_d[r];
                bi[row][colHalf] = best_i[r];
            }
        }
        __syncthreads();

        // ---- 3. combine halves, emit codes + histogram ------------------
        if (t < TILE_B) {
            float d0v = bd[t][0], d1v = bd[t][1];
            int   i0v = bi[t][0], i1v = bi[t][1];
            int code = (d1v < d0v || (d1v == d0v && i1v < i0v)) ? i1v : i0v;
            codes_sh[t] = code;
            out[OFF_CODES + (size_t)(blockRow0 + t) * R_N + s] = (float)code;
            atomicAdd(&counts[s * K_N + code], 1.0f);
        }
        __syncthreads();

        // ---- 4. exact fp32 residual update ------------------------------
        {
            const int code = codes_sh[rowT];
            const float* cp = embed + ((size_t)s * K_N + code) * D_N + q * 64;
            #pragma unroll
            for (int c = 0; c < 16; ++c) {
                v4f v = *(const v4f*)(cp + c * 4);
                r_reg[c * 4 + 0] -= v[0]; r_reg[c * 4 + 1] -= v[1];
                r_reg[c * 4 + 2] -= v[2]; r_reg[c * 4 + 3] -= v[3];
            }
        }
        __syncthreads();
    }

    // ---- epilogue: residual, z_q = z - residual, loss partial -----------
    {
        const size_t base = (size_t)(blockRow0 + rowT) * D_N + q * 64;
        const float* zp = z + base;
        float lsum = 0.0f;
        #pragma unroll
        for (int c = 0; c < 16; ++c) {
            v4f zv = *(const v4f*)(zp + c * 4);
            v4f rv, qv;
            #pragma unroll
            for (int j = 0; j < 4; ++j) {
                float r = r_reg[c * 4 + j];
                rv[j] = r; qv[j] = zv[j] - r; lsum += r * r;
            }
            *(v4f*)(out + OFF_RES + base + c * 4) = rv;
            *(v4f*)(out + OFF_ZQ  + base + c * 4) = qv;
        }
        #pragma unroll
        for (int off = 16; off >= 1; off >>= 1)
            lsum += shfl_xor_f(lsum, off, lane);
        if (lane == 0) atomicAdd(lossp, lsum);
    }
}

// ---------------------------------------------------------------------------
__global__ void rvq_final(const float* __restrict__ counts,
                          const float* __restrict__ lossp,
                          float* __restrict__ out) {
    int i = blockIdx.x * blockDim.x + threadIdx.x;
    if (i < R_N * K_N)
        out[OFF_USAGE + i] = counts[i] * (1.0f / (float)B_N);
    if (i == 0)
        out[OFF_LOSS] = lossp[0] * (1.0f / ((float)B_N * (float)D_N));
}

// ---------------------------------------------------------------------------
extern "C" void kernel_launch(void* const* d_in, const int* in_sizes, int n_in,
                              void* d_out, int out_size, void* d_ws, size_t ws_size,
                              hipStream_t stream) {
    const float* z     = (const float*)d_in[0];   // [B, D]
    const float* embed = (const float*)d_in[1];   // [R, K, D]

    // workspace layout: bf16 codebook (4 MB), e2 norms, counts, loss accum
    char*   ws     = (char*)d_ws;
    __bf16* cb16   = (__bf16*)ws;
    float*  e2     = (float*)(ws + (size_t)R_N * K_N * D_N * 2);
    float*  counts = e2 + R_N * K_N;
    float*  lossp  = counts + R_N * K_N;
    float*  out    = (float*)d_out;
    (void)in_sizes; (void)n_in; (void)out_size; (void)ws_size;

    rvq_zero<<<(R_N * K_N + 255) / 256, 256, 0, stream>>>(counts, lossp);
    rvq_prep_bf16<<<512, 256, 0, stream>>>(embed, cb16);
    rvq_prep_e2<<<(R_N * K_N + 255) / 256, 256, 0, stream>>>(embed, e2);
    rvq_main<<<B_N / TILE_B, THREADS, 0, stream>>>(z, embed, cb16, e2,
                                                   counts, lossp, out);
    rvq_final<<<(R_N * K_N + 255) / 256, 256, 0, stream>>>(counts, lossp, out);
}
